// LocalCorrModule_50379966382169
// MI455X (gfx1250) — matrix operand order repaired and checked by reading, not compiled, not run on hardware
//
#include <hip/hip_runtime.h>
#include <hip/hip_bf16.h>

// MI455X / gfx1250, wave32. WMMA used for the 64x64 channel Gram matrix
// C = corr^T * corr (K = 262144 spatial samples per batch), which lets us compute the
// InstanceNorm statistics of the spatial-attention branch analytically and
// saves a full 128MB pass over the corr tensor.

typedef __attribute__((ext_vector_type(16))) _Float16 v16h;
typedef __attribute__((ext_vector_type(8)))  _Float16 v8h;
typedef __attribute__((ext_vector_type(8)))  float    v8f;

union F16Frag { v16h v; v8h h[2]; };

#define EPSN   1e-5f
#define DHW    262144        // 64*64*64
#define INVN   (1.0f/262144.0f)
#define PADF   65            // f32 LDS row stride (64 ch + 1 pad -> conflict free)
#define STRT   88            // f16 [ch][tok] token stride: 176B = 11*16B aligned,
                             // 44-dword lane stride -> conflict-free b128 loads

__device__ __forceinline__ float sigm(float x) { return 1.f / (1.f + expf(-x)); }

// ---------------------------------------------------------------------------
// K1: LayerNorm(x), LayerNorm(y), windowed QK correlation -> corr (d_out),
//     plus gap partial sums (atomic) and per-block Gram partials via WMMA.
// Grid: 2048 blocks = (b=2, wd=32, wh=32); each block loops 4 slabs of
// 8 windows (16 w positions) => covers all 32 windows along w.
// ---------------------------------------------------------------------------
__global__ void __launch_bounds__(256)
k1_corr(const float* __restrict__ x_in, const float* __restrict__ y_in,
        const float* __restrict__ gxv, const float* __restrict__ bxv,
        const float* __restrict__ gyv, const float* __restrict__ byv,
        float* __restrict__ corr_out,      // == d_out  [2][64][64][64][64]
        float* __restrict__ gap,           // [2][64]  (pre-zeroed, atomic)
        float* __restrict__ c_part)        // [2048][4096] Gram partials
{
    __shared__ float xs[64 * PADF];        // 16640 B : x slab / corr f32
    __shared__ float ys[64 * PADF];        // 16640 B : y slab / corr f16 [ch][tok]
    __shared__ float prm[4 * 64];          // norm gamma/beta

    const int tid  = threadIdx.x;
    const int lane = tid & 31;
    const int wv   = tid >> 5;             // wave id 0..7
    const int blk  = blockIdx.x;
    const int b    = blk >> 10;
    const int wd   = (blk >> 5) & 31;
    const int wh   = blk & 31;

    if (tid < 64) {
        prm[tid]       = gxv[tid];
        prm[64 + tid]  = bxv[tid];
        prm[128 + tid] = gyv[tid];
        prm[192 + tid] = byv[tid];
    }

    float gacc = 0.f;                      // gap partial for channel==tid (tid<64)
    v8f acc0 = {};                         // Gram tile (ti0 , tj)
    v8f acc1 = {};                         // Gram tile (ti0+2, tj)
    const int ti0 = wv >> 2;               // 0..1
    const int tjc = wv & 3;                // 0..3
    const int lM  = lane & 15;
    const int lG  = lane >> 4;

    for (int it = 0; it < 4; ++it) {
        const int w0 = it * 16;
        __syncthreads();                   // protect LDS reuse across iterations

        // ---- load x/y slabs: [64c][2dz][2dy][16w], 1024 float4 each ----
        for (int i = 0; i < 4; ++i) {
            const int f   = tid + 256 * i;         // 0..1023
            const int row = f >> 2;                // (c,dz,dy)
            const int pos = f & 3;
            const int c   = row >> 2, dz = (row >> 1) & 1, dy = row & 1;
            const size_t ga = ((((size_t)(b * 64 + c) * 64 + (2 * wd + dz)) * 64
                               + (2 * wh + dy)) * 64) + w0 + pos * 4;
            const float4 vx = *(const float4*)(x_in + ga);
            const float4 vy = *(const float4*)(y_in + ga);
            const float* px = (const float*)&vx;
            const float* py = (const float*)&vy;
            const int tokb = dz * 32 + dy * 16 + pos * 4;
            #pragma unroll
            for (int j = 0; j < 4; ++j) {
                xs[(tokb + j) * PADF + c] = px[j];
                ys[(tokb + j) * PADF + c] = py[j];
            }
        }
        __syncthreads();

        // ---- per-token LayerNorm over 64 channels, in place ----
        if (tid < 128) {
            float* buf       = (tid < 64) ? xs : ys;
            const float* gg  = (tid < 64) ? prm        : prm + 128;
            const float* bb  = (tid < 64) ? prm + 64   : prm + 192;
            const int tok    = tid & 63;
            float s = 0.f;
            for (int c = 0; c < 64; ++c) s += buf[tok * PADF + c];
            const float mn = s * (1.f / 64.f);
            float v = 0.f;
            for (int c = 0; c < 64; ++c) {
                const float d = buf[tok * PADF + c] - mn;
                v = fmaf(d, d, v);
            }
            const float rstd = rsqrtf(v * (1.f / 64.f) + EPSN);
            for (int c = 0; c < 64; ++c)
                buf[tok * PADF + c] = fmaf((buf[tok * PADF + c] - mn) * rstd, gg[c], bb[c]);
        }
        __syncthreads();

        // ---- correlation: thread -> (voxel, 16 channels) ----
        // voxel tok' = tz*32+ty*16+wx ; head t = tz*4+ty*2+(wx&1) ; win = wx>>1
        // corr[tok'][n*8+m] = sum_d xn[tok(n)][t*8+d] * yn[tok(m)][t*8+d]
        const int vox  = tid >> 2;
        const int qq   = tid & 3;                 // n in {2qq, 2qq+1}
        const int tz   = vox >> 5, ty = (vox >> 4) & 1, wx = vox & 15;
        const int win  = wx >> 1;
        const int tpos = tz * 4 + ty * 2 + (wx & 1);
        float xr[2][8];
        #pragma unroll
        for (int n2 = 0; n2 < 2; ++n2) {
            const int n    = qq * 2 + n2;
            const int tokn = (n >> 2) * 32 + ((n >> 1) & 1) * 16 + win * 2 + (n & 1);
            #pragma unroll
            for (int d2 = 0; d2 < 8; ++d2) xr[n2][d2] = xs[tokn * PADF + tpos * 8 + d2];
        }
        float cres[16];
        #pragma unroll
        for (int m = 0; m < 8; ++m) {
            const int tokm = (m >> 2) * 32 + ((m >> 1) & 1) * 16 + win * 2 + (m & 1);
            float yv[8];
            #pragma unroll
            for (int d2 = 0; d2 < 8; ++d2) yv[d2] = ys[tokm * PADF + tpos * 8 + d2];
            #pragma unroll
            for (int n2 = 0; n2 < 2; ++n2) {
                float s = 0.f;
                #pragma unroll
                for (int d2 = 0; d2 < 8; ++d2) s = fmaf(xr[n2][d2], yv[d2], s);
                cres[n2 * 8 + m] = s;
            }
        }
        __syncthreads();                   // all reads of xn/yn done

        // ---- stash corr: f32 [vox][ch] into xs, f16 [ch][tok] into ys ----
        _Float16* ch16 = (_Float16*)ys;
        #pragma unroll
        for (int n2 = 0; n2 < 2; ++n2)
            #pragma unroll
            for (int m = 0; m < 8; ++m) {
                const int ch = (qq * 2 + n2) * 8 + m;
                xs[vox * PADF + ch]  = cres[n2 * 8 + m];
                ch16[ch * STRT + vox] = (_Float16)cres[n2 * 8 + m];
            }
        __syncthreads();

        // ---- gap partial ----
        if (tid < 64)
            for (int tok = 0; tok < 64; ++tok) gacc += xs[tok * PADF + tid];

        // ---- coalesced global store of corr ----
        for (int i = 0; i < 4; ++i) {
            const int f   = tid + 256 * i;
            const int row = f >> 2;
            const int pos = f & 3;
            const int c   = row >> 2, dz = (row >> 1) & 1, dy = row & 1;
            float o[4];
            #pragma unroll
            for (int j = 0; j < 4; ++j)
                o[j] = xs[(dz * 32 + dy * 16 + pos * 4 + j) * PADF + c];
            const size_t ga = ((((size_t)(b * 64 + c) * 64 + (2 * wd + dz)) * 64
                               + (2 * wh + dy)) * 64) + w0 + pos * 4;
            *(float4*)(corr_out + ga) = *(float4*)o;
        }

        // ---- WMMA Gram accumulation: C_tile += A(16x32) * B(32x16) ----
        // A[M,K] = corr16[ch=M][tok=K], B[K,N] = corr16[ch=N][tok=K].
        // With the [ch][tok] layout and the ISA 16-bit fragment layouts,
        // each lane's fragment halves are contiguous tokens:
        //   A: elems 0-7  = K 8*lG+0..7      (one 16B ds_load_b128)
        //      elems 8-15 = K 16+8*lG+0..7   (one 16B ds_load_b128)
        //   B: same as A (K 8*lG+0..7, 16+8*lG+0..7; two 16B ds_load_b128)
        #pragma unroll
        for (int kc = 0; kc < 2; ++kc) {
            F16Frag A0, A1, Bm;
            const _Float16* arow0 = ch16 + (ti0 * 16 + lM)       * STRT + kc * 32 + lG * 8;
            const _Float16* arow1 = ch16 + ((ti0 + 2) * 16 + lM) * STRT + kc * 32 + lG * 8;
            const _Float16* brow  = ch16 + (tjc * 16 + lM)       * STRT + kc * 32 + lG * 8;
            A0.h[0] = *(const v8h*)(arow0);
            A0.h[1] = *(const v8h*)(arow0 + 16);
            A1.h[0] = *(const v8h*)(arow1);
            A1.h[1] = *(const v8h*)(arow1 + 16);
            Bm.h[0] = *(const v8h*)(brow);
            Bm.h[1] = *(const v8h*)(brow + 16);
            acc0 = __builtin_amdgcn_wmma_f32_16x16x32_f16(false, A0.v, false, Bm.v,
                                                          (short)0, acc0, false, false);
            acc1 = __builtin_amdgcn_wmma_f32_16x16x32_f16(false, A1.v, false, Bm.v,
                                                          (short)0, acc1, false, false);
        }
    }

    if (tid < 64) atomicAdd(&gap[b * 64 + tid], gacc);

    // D layout (32-bit C/D 16x16): lane L: N = L%16, M = r + 8*(L/16)
    float* cp = c_part + (size_t)blk * 4096;
    #pragma unroll
    for (int r = 0; r < 8; ++r) {
        const int M = ti0 * 16 + lG * 8 + r;
        const int N = tjc * 16 + lM;
        cp[M * 64 + N]        = acc0[r];
        cp[(M + 32) * 64 + N] = acc1[r];
    }
}

// ---------------------------------------------------------------------------
// K2: reduce Gram partials over the 1024 blocks of each batch -> Cmean
// ---------------------------------------------------------------------------
__global__ void __launch_bounds__(256)
k2_reduce(const float* __restrict__ c_part, float* __restrict__ c_mean)
{
    const int idx = blockIdx.x * 256 + threadIdx.x;   // 0..8191
    const int b = idx >> 12, e = idx & 4095;
    float s = 0.f;
    for (int p = 0; p < 1024; ++p)
        s += c_part[(size_t)(b * 1024 + p) * 4096 + e];
    c_mean[idx] = s * INVN;
}

// ---------------------------------------------------------------------------
// K3: channel-attention gate a, u = a .* sa_w1, analytic InstanceNorm stats.
// Grid: (2 batches) x 64 threads.
// ---------------------------------------------------------------------------
__global__ void __launch_bounds__(64)
k3_stats(const float* __restrict__ gap, const float* __restrict__ cmean,
         const float* __restrict__ ca_w1, const float* __restrict__ ca_b1,
         const float* __restrict__ ca_w2, const float* __restrict__ ca_b2,
         const float* __restrict__ sa_w1, const float* __restrict__ sa_b1,
         float* __restrict__ a_out, float* __restrict__ u_out,
         float* __restrict__ m_out, float* __restrict__ rstd_out)
{
    const int b = blockIdx.x, t = threadIdx.x;
    __shared__ float gm[64], hh[4], uu[256];
    gm[t] = gap[b * 64 + t] * INVN;
    __syncthreads();
    if (t < 4) {
        float s = ca_b1[t];
        for (int c = 0; c < 64; ++c) s = fmaf(ca_w1[t * 64 + c], gm[c], s);
        hh[t] = fmaxf(s, 0.f);
    }
    __syncthreads();
    float s2 = ca_b2[t];
    #pragma unroll
    for (int o = 0; o < 4; ++o) s2 = fmaf(ca_w2[t * 4 + o], hh[o], s2);
    const float av = sigm(s2);
    a_out[b * 64 + t] = av;
    #pragma unroll
    for (int cr = 0; cr < 4; ++cr) {
        const float uv = av * sa_w1[cr * 64 + t];
        uu[cr * 64 + t] = uv;
        u_out[(b * 4 + cr) * 64 + t] = uv;
    }
    __syncthreads();
    if (t < 4) {
        const float b1 = sa_b1[t];
        float mz = 0.f;
        for (int c = 0; c < 64; ++c) mz = fmaf(uu[t * 64 + c], gm[c], mz);
        float q = 0.f;
        for (int c = 0; c < 64; ++c) {
            float inner = 0.f;
            for (int c2 = 0; c2 < 64; ++c2)
                inner = fmaf(uu[t * 64 + c2], cmean[b * 4096 + c * 64 + c2], inner);
            q = fmaf(uu[t * 64 + c], inner, q);
        }
        const float m   = mz + b1;
        const float esq = q + 2.f * b1 * mz + b1 * b1;
        const float var = esq - m * m;
        m_out[b * 4 + t]    = m;
        rstd_out[b * 4 + t] = rsqrtf(fmaxf(var, 0.f) + EPSN);
    }
}

// ---------------------------------------------------------------------------
// K4: in-place gating of d_out: out = sigmoid(w2.relu((u.corr+b1-m)*rstd)+b2)
//     * a .* corr.  Grid: 4096 blocks x 128 threads, 128 voxels per block.
// ---------------------------------------------------------------------------
__global__ void __launch_bounds__(128)
k4_final(float* __restrict__ io,
         const float* __restrict__ a_in, const float* __restrict__ u_in,
         const float* __restrict__ m_in, const float* __restrict__ rstd_in,
         const float* __restrict__ sa_b1, const float* __restrict__ sa_w2,
         const float* __restrict__ sa_b2)
{
    __shared__ float tile[128 * PADF];
    __shared__ float aa[64], uu[256], ms[4], rs[4], b1s[4], w2s[4], b2s;
    const int tid = threadIdx.x;
    const int blk = blockIdx.x;
    const int b   = blk >> 11;
    const size_t v0 = (size_t)(blk & 2047) * 128;

    if (tid < 64) {
        aa[tid] = a_in[b * 64 + tid];
        #pragma unroll
        for (int cr = 0; cr < 4; ++cr) uu[cr * 64 + tid] = u_in[(b * 4 + cr) * 64 + tid];
    } else if (tid < 68) {
        const int cr = tid - 64;
        ms[cr]  = m_in[b * 4 + cr];
        rs[cr]  = rstd_in[b * 4 + cr];
        b1s[cr] = sa_b1[cr];
        w2s[cr] = sa_w2[cr];
        if (cr == 0) b2s = sa_b2[0];
    }
    // load 64ch x 128vox tile (coalesced), then barrier before in-place write
    for (int i = 0; i < 16; ++i) {
        const int f = tid + 128 * i;              // 0..2047 float4s
        const int c = f >> 5, pos = f & 31;
        const float4 vv = *(const float4*)(io + (size_t)(b * 64 + c) * DHW + v0 + pos * 4);
        const float* vp = (const float*)&vv;
        #pragma unroll
        for (int j = 0; j < 4; ++j) tile[(pos * 4 + j) * PADF + c] = vp[j];
    }
    __syncthreads();

    float accv[4] = {0.f, 0.f, 0.f, 0.f};
    for (int c = 0; c < 64; ++c) {
        const float vv = tile[tid * PADF + c];
        #pragma unroll
        for (int cr = 0; cr < 4; ++cr) accv[cr] = fmaf(uu[cr * 64 + c], vv, accv[cr]);
    }
    float gsum = b2s;
    #pragma unroll
    for (int cr = 0; cr < 4; ++cr) {
        const float sn = (accv[cr] + b1s[cr] - ms[cr]) * rs[cr];
        gsum = fmaf(w2s[cr], fmaxf(sn, 0.f), gsum);
    }
    const float g = sigm(gsum);
    for (int c = 0; c < 64; ++c)
        io[(size_t)(b * 64 + c) * DHW + v0 + tid] = g * aa[c] * tile[tid * PADF + c];
}

// ---------------------------------------------------------------------------
// ws layout (floats): [0,128) gap | [128,256) a | [256,768) u |
// [768,776) m | [776,784) rstd | [1024,9216) Cmean | [16384, +8388608) Cpart
// total ~33.6 MB.  gap is zeroed each launch (atomic accumulators); all other
// regions are fully overwritten before use -> deterministic across replays.
// ---------------------------------------------------------------------------
extern "C" void kernel_launch(void* const* d_in, const int* in_sizes, int n_in,
                              void* d_out, int out_size, void* d_ws, size_t ws_size,
                              hipStream_t stream) {
    const float* x_in   = (const float*)d_in[0];
    const float* y_in   = (const float*)d_in[1];
    const float* normxg = (const float*)d_in[2];
    const float* normxb = (const float*)d_in[3];
    const float* normyg = (const float*)d_in[4];
    const float* normyb = (const float*)d_in[5];
    const float* ca_w1  = (const float*)d_in[6];
    const float* ca_b1  = (const float*)d_in[7];
    const float* ca_w2  = (const float*)d_in[8];
    const float* ca_b2  = (const float*)d_in[9];
    const float* sa_w1  = (const float*)d_in[10];
    const float* sa_b1  = (const float*)d_in[11];
    const float* sa_w2  = (const float*)d_in[12];
    const float* sa_b2  = (const float*)d_in[13];
    float* out = (float*)d_out;
    float* ws  = (float*)d_ws;

    float* gap    = ws;
    float* a_buf  = ws + 128;
    float* u_buf  = ws + 256;
    float* m_buf  = ws + 768;
    float* r_buf  = ws + 776;
    float* cmean  = ws + 1024;
    float* cpart  = ws + 16384;

    hipMemsetAsync(gap, 0, 128 * sizeof(float), stream);

    k1_corr<<<2048, 256, 0, stream>>>(x_in, y_in, normxg, normxb, normyg, normyb,
                                      out, gap, cpart);
    k2_reduce<<<32, 256, 0, stream>>>(cpart, cmean);
    k3_stats<<<2, 64, 0, stream>>>(gap, cmean, ca_w1, ca_b1, ca_w2, ca_b2,
                                   sa_w1, sa_b1, a_buf, u_buf, m_buf, r_buf);
    k4_final<<<4096, 128, 0, stream>>>(out, a_buf, u_buf, m_buf, r_buf,
                                       sa_b1, sa_w2, sa_b2);
}
